// CausalMultiheadSelfAttention_21895743275638
// MI455X (gfx1250) — matrix operand order repaired and checked by reading, not compiled, not run on hardware
//
#include <hip/hip_runtime.h>
#include <hip/hip_bf16.h>

#define SEQ   8192
#define BATCH 4
#define EMBED 1024
#define HEADS 16
#define WIN   256
#define HDIM  64

typedef __attribute__((ext_vector_type(16))) __bf16 v16bf;
typedef __attribute__((ext_vector_type(2)))  __bf16 v2bf;
typedef __attribute__((ext_vector_type(8)))  float  v8f;
typedef __attribute__((ext_vector_type(4)))  int    v4i;

union Frag { v16bf v; uint4 q[2]; };
union Acc  { v8f   v; float f[8]; };

// Native bf16 converts -> v_cvt_pk_bf16_f32 on gfx1250
__device__ __forceinline__ unsigned short f2bf(float x) {
  union { __bf16 h; unsigned short u; } c; c.h = (__bf16)x;
  return c.u;
}
__device__ __forceinline__ unsigned int pk_bf16(float lo, float hi) {
  union { v2bf v; unsigned int u; } c;
  c.v.x = (__bf16)lo; c.v.y = (__bf16)hi;
  return c.u;
}
__device__ __forceinline__ float bf2f(unsigned short h) {
  union { unsigned short u; __bf16 h; } c; c.u = h;
  return (float)c.h;
}

// gfx1250 async global->LDS copy (ASYNCcnt-tracked, no VGPR round-trip).
#if __has_builtin(__builtin_amdgcn_global_load_async_to_lds_b128) && \
    __has_builtin(__builtin_amdgcn_s_wait_asynccnt)
#define HAVE_ASYNC_LDS 1
__device__ __forceinline__ void async_copy_b128(void* lds, const void* g) {
  __builtin_amdgcn_global_load_async_to_lds_b128(
      (__attribute__((address_space(1))) v4i*)(g),
      (__attribute__((address_space(3))) v4i*)(lds), 0, 0);
}
__device__ __forceinline__ void async_fence() {
  __builtin_amdgcn_s_wait_asynccnt(0);
}
#else
#define HAVE_ASYNC_LDS 0
__device__ __forceinline__ void async_copy_b128(void* lds, const void* g) {
  *(uint4*)lds = *(const uint4*)g;
}
__device__ __forceinline__ void async_fence() {}
#endif

// ---------------------------------------------------------------------------
// Kernel 1: C[bs, n] = (x gathered as (B*S, E)) @ W + bias, output bf16.
// Tile 128x128x32, double-buffered LDS, software-pipelined staging.
// ---------------------------------------------------------------------------
__global__ __launch_bounds__(256) void qkv_gemm_kernel(
    const float* __restrict__ X, const float* __restrict__ W,
    const float* __restrict__ bias, unsigned short* __restrict__ out, int Ntot)
{
  __shared__ unsigned short ldsA[2][128][40];   // bf16, padded rows
  __shared__ unsigned short ldsB[2][128][40];   // stored transposed: [n][k]
  const int tid   = threadIdx.x;
  const int lane  = tid & 31;
  const int wave  = tid >> 5;
  const int wm    = wave >> 2, wn = wave & 3;
  const int nlane = lane & 15;
  const int khalf = (lane >> 4) * 8;
  const int kb16  = (lane >> 4) * 16;
  const int m0 = blockIdx.y * 128;
  const int n0 = blockIdx.x * 128;

  Acc acc[4][2];
#pragma unroll
  for (int a = 0; a < 4; ++a)
#pragma unroll
    for (int c = 0; c < 2; ++c)
#pragma unroll
      for (int r = 0; r < 8; ++r) acc[a][c].f[r] = 0.f;

  float4 fA[4], fB[4];

  auto loadA = [&](int k0) {     // global -> regs (no waits yet)
#pragma unroll
    for (int j = 0; j < 4; ++j) {
      int flat = tid + j * 256;
      int r = flat >> 3, kq = flat & 7;
      int m = m0 + r;
      int s = m & (SEQ - 1), bb = m >> 13;
      fA[j] = *(const float4*)&X[((size_t)s * BATCH + bb) * EMBED + k0 + kq * 4];
    }
  };
  auto loadB = [&](int k0) {
#pragma unroll
    for (int j = 0; j < 4; ++j) {
      int flat = tid + j * 256;
      int kk = flat >> 5, nq = flat & 31;
      fB[j] = *(const float4*)&W[(size_t)(k0 + kk) * Ntot + n0 + nq * 4];
    }
    if (k0 + 32 < EMBED)  // global_prefetch_b8 one tile further out
      __builtin_prefetch(&W[(size_t)(k0 + 32) * Ntot + n0 + lane * 4], 0, 0);
  };
  auto store = [&](int buf) {    // cvt + LDS stores (waits land here)
#pragma unroll
    for (int j = 0; j < 4; ++j) {
      int flat = tid + j * 256;
      int r = flat >> 3, kq = flat & 7;
      *(uint2*)&ldsA[buf][r][kq * 4] =
          make_uint2(pk_bf16(fA[j].x, fA[j].y), pk_bf16(fA[j].z, fA[j].w));
    }
#pragma unroll
    for (int j = 0; j < 4; ++j) {
      int flat = tid + j * 256;
      int kk = flat >> 5, nq = flat & 31;
      ldsB[buf][nq * 4 + 0][kk] = f2bf(fB[j].x);
      ldsB[buf][nq * 4 + 1][kk] = f2bf(fB[j].y);
      ldsB[buf][nq * 4 + 2][kk] = f2bf(fB[j].z);
      ldsB[buf][nq * 4 + 3][kk] = f2bf(fB[j].w);
    }
  };
  auto compute = [&](int buf) {
    Frag fa[4], fb[2];
#pragma unroll
    for (int mt = 0; mt < 4; ++mt) {  // A-frag: m=lane&15, k=khalf+i / 16+khalf+i
      const int row = wm * 64 + mt * 16 + nlane;
      fa[mt].q[0] = *(const uint4*)&ldsA[buf][row][khalf];
      fa[mt].q[1] = *(const uint4*)&ldsA[buf][row][16 + khalf];
    }
#pragma unroll
    for (int nt = 0; nt < 2; ++nt) {  // B-frag: n=lane&15, k=khalf+i / 16+khalf+i
      const int col = wn * 32 + nt * 16 + nlane;
      fb[nt].q[0] = *(const uint4*)&ldsB[buf][col][khalf];
      fb[nt].q[1] = *(const uint4*)&ldsB[buf][col][16 + khalf];
    }
#pragma unroll
    for (int mt = 0; mt < 4; ++mt)
#pragma unroll
      for (int nt = 0; nt < 2; ++nt)
        acc[mt][nt].v = __builtin_amdgcn_wmma_f32_16x16x32_bf16(
            false, fa[mt].v, false, fb[nt].v, (short)0, acc[mt][nt].v, false, false);
  };

  loadA(0); loadB(0); store(0);
  __syncthreads();
  for (int kt = 0; kt < 32; kt += 2) {
    loadA((kt + 1) * 32); loadB((kt + 1) * 32);   // issue next tile's loads
    compute(0);                                   // WMMA hides load latency
    store(1);
    __syncthreads();
    if (kt + 2 < 32) { loadA((kt + 2) * 32); loadB((kt + 2) * 32); }
    compute(1);
    if (kt + 2 < 32) store(0);
    __syncthreads();
  }

#pragma unroll
  for (int mt = 0; mt < 4; ++mt)
#pragma unroll
    for (int nt = 0; nt < 2; ++nt) {
      const int col = n0 + wn * 32 + nt * 16 + nlane;
      const float bv = bias[col];
#pragma unroll
      for (int r = 0; r < 8; ++r) {  // C layout: M = r + 8*(lane>=16), N = lane&15
        const int row = m0 + wm * 64 + mt * 16 + r + ((lane >> 4) << 3);
        out[(size_t)row * Ntot + col] = f2bf(acc[mt][nt].f[r] + bv);
      }
    }
}

// ---------------------------------------------------------------------------
// Kernel 2: in-place RoPE on bf16 rows, pos = WIN + s, half = 32.
// ---------------------------------------------------------------------------
__global__ __launch_bounds__(256) void rope_kernel(unsigned short* __restrict__ buf, int ld)
{
  const int idx = blockIdx.x * 256 + threadIdx.x;   // (row, h, d<32)
  const int m = idx >> 9;
  const int p = idx & 511;
  const int h = p >> 5, d = p & 31;
  const int s = m & (SEQ - 1);
  const float pos = (float)(WIN + s);
  const float inv = __expf(-(float)d * (9.210340371976184f / 32.0f)); // 10000^{-d/32}
  float sn, cs;
  __sincosf(pos * inv, &sn, &cs);
  unsigned short* row = buf + (size_t)m * ld + h * HDIM;
  const float x1 = bf2f(row[d]);
  const float x2 = bf2f(row[d + 32]);
  row[d]      = f2bf(x1 * cs - x2 * sn);
  row[d + 32] = f2bf(x1 * sn + x2 * cs);
}

// ---------------------------------------------------------------------------
// Kernel 3: sliding-window attention with ALiBi, flash-style online softmax.
// One block per (window-block n, head h, batch b); 8 waves x 32 query rows.
// ---------------------------------------------------------------------------
__global__ __launch_bounds__(256) void swa_attn_kernel(
    const unsigned short* __restrict__ Q, const unsigned short* __restrict__ KV,
    unsigned short* __restrict__ O)
{
  __shared__ unsigned short ldsK[64][72];      // key chunk, row-major [key][d]
  __shared__ unsigned short ldsVT[64][72];     // value chunk, transposed [d][key]
  __shared__ unsigned short ldsP[8][32][72];   // per-wave P staging (C->A relayout)
  const int tid   = threadIdx.x;
  const int lane  = tid & 31, wave = tid >> 5;
  const int nlane = lane & 15;
  const int hi8   = (lane >> 4) << 3;
  const int khalf = (lane >> 4) * 8;
  const int kb16  = (lane >> 4) * 16;
  const int nblk = blockIdx.x, h = blockIdx.y, b = blockIdx.z;
  const float slope = exp2f(-8.0f * (float)(h + 1) / (float)HEADS);

  // Resident Q A-fragments (direct global loads, rows contiguous in d)
  Frag fq[2][2];
#pragma unroll
  for (int mt = 0; mt < 2; ++mt)
#pragma unroll
    for (int ks = 0; ks < 2; ++ks) {
      const int qrow = b * SEQ + nblk * WIN + wave * 32 + mt * 16 + nlane;
      const unsigned short* base = Q + (size_t)qrow * EMBED + h * HDIM + ks * 32;
      fq[mt][ks].q[0] = *(const uint4*)(base + khalf);
      fq[mt][ks].q[1] = *(const uint4*)(base + 16 + khalf);
    }

  Acc oacc[2][4];
  float m_i[2][8], l_i[2][8];
#pragma unroll
  for (int mt = 0; mt < 2; ++mt)
#pragma unroll
    for (int r = 0; r < 8; ++r) { m_i[mt][r] = -1e30f; l_i[mt][r] = 0.f; }
#pragma unroll
  for (int mt = 0; mt < 2; ++mt)
#pragma unroll
    for (int nt = 0; nt < 4; ++nt)
#pragma unroll
      for (int r = 0; r < 8; ++r) oacc[mt][nt].f[r] = 0.f;

  for (int chunk = 0; chunk < 8; ++chunk) {
    // Stage K chunk (64 keys x 64 dims, bf16): async global->LDS where valid,
    // zero-fill the left window padding.
#pragma unroll
    for (int t = 0; t < 2; ++t) {
      int flat = tid + t * 256;
      int kr = flat >> 3, q8 = flat & 7;
      int pos = nblk * WIN + chunk * 64 + kr - WIN;
      unsigned short* dst = &ldsK[kr][q8 * 8];
      if (pos >= 0)
        async_copy_b128(dst, &KV[((size_t)(b * SEQ + pos)) * 2048 + h * HDIM + q8 * 8]);
      else
        *(uint4*)dst = make_uint4(0u, 0u, 0u, 0u);
    }
    // Stage V chunk transposed: load batch into regs, then scatter-store
    unsigned int vv[8];
#pragma unroll
    for (int t = 0; t < 8; ++t) {
      int flat = tid + t * 256;
      int kr = flat >> 5, dp = flat & 31;
      int pos = nblk * WIN + chunk * 64 + kr - WIN;
      vv[t] = 0u;
      if (pos >= 0)
        vv[t] = *(const unsigned int*)&KV[(size_t)(b * SEQ + pos) * 2048 + 1024 + h * HDIM + dp * 2];
    }
#pragma unroll
    for (int t = 0; t < 8; ++t) {
      int flat = tid + t * 256;
      int kr = flat >> 5, dp = flat & 31;
      ldsVT[dp * 2 + 0][kr] = (unsigned short)(vv[t] & 0xFFFFu);
      ldsVT[dp * 2 + 1][kr] = (unsigned short)(vv[t] >> 16);
    }
    async_fence();       // s_wait_asynccnt 0
    __syncthreads();

    // Scores S = (Q K^T) over D=64 (two k-steps of 32)
    Acc sc[2][4];
#pragma unroll
    for (int nt = 0; nt < 4; ++nt) {
      Frag fk[2];
      const int key = nt * 16 + nlane;
#pragma unroll
      for (int ks = 0; ks < 2; ++ks) {
        fk[ks].q[0] = *(const uint4*)&ldsK[key][ks * 32 + khalf];
        fk[ks].q[1] = *(const uint4*)&ldsK[key][ks * 32 + 16 + khalf];
      }
#pragma unroll
      for (int mt = 0; mt < 2; ++mt) {
        Acc c;
#pragma unroll
        for (int r = 0; r < 8; ++r) c.f[r] = 0.f;
#pragma unroll
        for (int ks = 0; ks < 2; ++ks)
          c.v = __builtin_amdgcn_wmma_f32_16x16x32_bf16(
              false, fq[mt][ks].v, false, fk[ks].v, (short)0, c.v, false, false);
        sc[mt][nt] = c;
      }
    }

    // Mask + ALiBi + online softmax (16-lane halves reduce independently)
#pragma unroll
    for (int mt = 0; mt < 2; ++mt)
#pragma unroll
      for (int r = 0; r < 8; ++r) {
        const int i = wave * 32 + mt * 16 + r + hi8;   // query idx in block
        float rmax = -1e30f;
#pragma unroll
        for (int nt = 0; nt < 4; ++nt) {
          const int j = chunk * 64 + nt * 16 + nlane;  // key idx in window
          const int rel = i - j + WIN;
          const bool ok = (rel >= 0) && (rel <= WIN) && (nblk * WIN + j >= WIN);
          const float sv = ok ? sc[mt][nt].f[r] * 0.125f - slope * (float)rel : -1e30f;
          sc[mt][nt].f[r] = sv;
          rmax = fmaxf(rmax, sv);
        }
        rmax = fmaxf(rmax, __shfl_xor(rmax, 1, 32));
        rmax = fmaxf(rmax, __shfl_xor(rmax, 2, 32));
        rmax = fmaxf(rmax, __shfl_xor(rmax, 4, 32));
        rmax = fmaxf(rmax, __shfl_xor(rmax, 8, 32));
        const float mnew = fmaxf(m_i[mt][r], rmax);
        const float scal = __expf(m_i[mt][r] - mnew);
        m_i[mt][r] = mnew;
        float rs = 0.f;
#pragma unroll
        for (int nt = 0; nt < 4; ++nt) {
          const float pv = __expf(sc[mt][nt].f[r] - mnew);
          rs += pv;
          ldsP[wave][mt * 16 + r + hi8][nt * 16 + nlane] = f2bf(pv);
          oacc[mt][nt].f[r] *= scal;
        }
        rs += __shfl_xor(rs, 1, 32);
        rs += __shfl_xor(rs, 2, 32);
        rs += __shfl_xor(rs, 4, 32);
        rs += __shfl_xor(rs, 8, 32);
        l_i[mt][r] = l_i[mt][r] * scal + rs;
      }

    // O += P @ V  (P A-frags from per-wave LDS, V B-frags from transposed LDS)
    Frag fp[2][2];
#pragma unroll
    for (int mt = 0; mt < 2; ++mt)
#pragma unroll
      for (int ks = 0; ks < 2; ++ks) {
        const unsigned short* pr = &ldsP[wave][mt * 16 + nlane][ks * 32];
        fp[mt][ks].q[0] = *(const uint4*)(pr + khalf);
        fp[mt][ks].q[1] = *(const uint4*)(pr + 16 + khalf);
      }
#pragma unroll
    for (int nt = 0; nt < 4; ++nt) {
      Frag fv[2];
      const int drow = nt * 16 + nlane;
#pragma unroll
      for (int ks = 0; ks < 2; ++ks) {
        fv[ks].q[0] = *(const uint4*)&ldsVT[drow][ks * 32 + khalf];
        fv[ks].q[1] = *(const uint4*)&ldsVT[drow][ks * 32 + 16 + khalf];
      }
#pragma unroll
      for (int mt = 0; mt < 2; ++mt)
#pragma unroll
        for (int ks = 0; ks < 2; ++ks)
          oacc[mt][nt].v = __builtin_amdgcn_wmma_f32_16x16x32_bf16(
              false, fp[mt][ks].v, false, fv[ks].v, (short)0, oacc[mt][nt].v, false, false);
    }
    __syncthreads();
  }

  // Normalize and write bf16 attention output (B*S, E) layout
#pragma unroll
  for (int mt = 0; mt < 2; ++mt)
#pragma unroll
    for (int r = 0; r < 8; ++r) {
      const float invl = 1.0f / l_i[mt][r];
      const int qrow = b * SEQ + nblk * WIN + wave * 32 + mt * 16 + r + hi8;
#pragma unroll
      for (int nt = 0; nt < 4; ++nt) {
        const int col = h * HDIM + nt * 16 + nlane;
        O[(size_t)qrow * EMBED + col] = f2bf(oacc[mt][nt].f[r] * invl);
      }
    }
}

// ---------------------------------------------------------------------------
// Kernel 4: out = (attn bf16) @ Wo + bo, scattered to (S,B,E) fp32.
// Double-buffered; A tile staged via async global->LDS copies.
// ---------------------------------------------------------------------------
__global__ __launch_bounds__(256) void out_proj_kernel(
    const unsigned short* __restrict__ A, const float* __restrict__ W,
    const float* __restrict__ bias, float* __restrict__ out)
{
  __shared__ unsigned short ldsA[2][128][40];
  __shared__ unsigned short ldsB[2][128][40];
  const int tid   = threadIdx.x;
  const int lane  = tid & 31;
  const int wave  = tid >> 5;
  const int wm    = wave >> 2, wn = wave & 3;
  const int nlane = lane & 15;
  const int khalf = (lane >> 4) * 8;
  const int kb16  = (lane >> 4) * 16;
  const int m0 = blockIdx.y * 128;
  const int n0 = blockIdx.x * 128;

  Acc acc[4][2];
#pragma unroll
  for (int a = 0; a < 4; ++a)
#pragma unroll
    for (int c = 0; c < 2; ++c)
#pragma unroll
      for (int r = 0; r < 8; ++r) acc[a][c].f[r] = 0.f;

  float4 fB[4];

  auto stageA = [&](int buf, int k0) {   // async b128 copies, no VGPR round-trip
#pragma unroll
    for (int j = 0; j < 2; ++j) {
      int flat = tid + j * 256;
      int r = flat >> 2, q8 = flat & 3;
      async_copy_b128(&ldsA[buf][r][q8 * 8],
                      &A[(size_t)(m0 + r) * EMBED + k0 + q8 * 8]);
    }
  };
  auto loadB = [&](int k0) {
#pragma unroll
    for (int j = 0; j < 4; ++j) {
      int flat = tid + j * 256;
      int kk = flat >> 5, nq = flat & 31;
      fB[j] = *(const float4*)&W[(size_t)(k0 + kk) * EMBED + n0 + nq * 4];
    }
    if (k0 + 32 < EMBED)
      __builtin_prefetch(&W[(size_t)(k0 + 32) * EMBED + n0 + lane * 4], 0, 0);
  };
  auto storeB = [&](int buf) {
#pragma unroll
    for (int j = 0; j < 4; ++j) {
      int flat = tid + j * 256;
      int kk = flat >> 5, nq = flat & 31;
      ldsB[buf][nq * 4 + 0][kk] = f2bf(fB[j].x);
      ldsB[buf][nq * 4 + 1][kk] = f2bf(fB[j].y);
      ldsB[buf][nq * 4 + 2][kk] = f2bf(fB[j].z);
      ldsB[buf][nq * 4 + 3][kk] = f2bf(fB[j].w);
    }
  };
  auto compute = [&](int buf) {
    Frag fa[4], fb[2];
#pragma unroll
    for (int mt = 0; mt < 4; ++mt) {
      const int row = wm * 64 + mt * 16 + nlane;
      fa[mt].q[0] = *(const uint4*)&ldsA[buf][row][khalf];
      fa[mt].q[1] = *(const uint4*)&ldsA[buf][row][16 + khalf];
    }
#pragma unroll
    for (int nt = 0; nt < 2; ++nt) {
      const int col = wn * 32 + nt * 16 + nlane;
      fb[nt].q[0] = *(const uint4*)&ldsB[buf][col][khalf];
      fb[nt].q[1] = *(const uint4*)&ldsB[buf][col][16 + khalf];
    }
#pragma unroll
    for (int mt = 0; mt < 4; ++mt)
#pragma unroll
      for (int nt = 0; nt < 2; ++nt)
        acc[mt][nt].v = __builtin_amdgcn_wmma_f32_16x16x32_bf16(
            false, fa[mt].v, false, fb[nt].v, (short)0, acc[mt][nt].v, false, false);
  };

  stageA(0, 0); loadB(0); storeB(0);
  async_fence();
  __syncthreads();
  for (int kt = 0; kt < 32; kt += 2) {
    stageA(1, (kt + 1) * 32); loadB((kt + 1) * 32);
    compute(0);
    storeB(1);
    async_fence();
    __syncthreads();
    if (kt + 2 < 32) { stageA(0, (kt + 2) * 32); loadB((kt + 2) * 32); }
    compute(1);
    if (kt + 2 < 32) storeB(0);
    async_fence();
    __syncthreads();
  }

#pragma unroll
  for (int mt = 0; mt < 4; ++mt)
#pragma unroll
    for (int nt = 0; nt < 2; ++nt) {
      const int col = n0 + wn * 32 + nt * 16 + nlane;
      const float bv = bias[col];
#pragma unroll
      for (int r = 0; r < 8; ++r) {
        const int m = m0 + wm * 64 + mt * 16 + r + ((lane >> 4) << 3);
        const int s = m & (SEQ - 1), bb = m >> 13;  // (B,S) -> (S,B) transpose
        out[((size_t)s * BATCH + bb) * EMBED + col] = acc[mt][nt].f[r] + bv;
      }
    }
}

// ---------------------------------------------------------------------------
extern "C" void kernel_launch(void* const* d_in, const int* in_sizes, int n_in,
                              void* d_out, int out_size, void* d_ws, size_t ws_size,
                              hipStream_t stream) {
  (void)in_sizes; (void)n_in; (void)out_size; (void)ws_size;
  const float* x   = (const float*)d_in[0];
  const float* Wq  = (const float*)d_in[1];
  const float* bq  = (const float*)d_in[2];
  const float* Wkv = (const float*)d_in[3];
  const float* bkv = (const float*)d_in[4];
  const float* Wo  = (const float*)d_in[5];
  const float* bo  = (const float*)d_in[6];
  float* out = (float*)d_out;

  const size_t MS = (size_t)BATCH * SEQ;                 // 32768 rows
  unsigned short* Qb  = (unsigned short*)d_ws;           // (B*S, 1024) bf16
  unsigned short* KVb = Qb + MS * EMBED;                 // (B*S, 2048) bf16: [K | V]
  unsigned short* Ob  = KVb + MS * 2 * EMBED;            // (B*S, 1024) bf16

  dim3 blk(256);
  qkv_gemm_kernel<<<dim3(EMBED / 128, MS / 128), blk, 0, stream>>>(x, Wq, bq, Qb, EMBED);
  qkv_gemm_kernel<<<dim3(2 * EMBED / 128, MS / 128), blk, 0, stream>>>(x, Wkv, bkv, KVb, 2 * EMBED);

  const int rope_threads = (int)(MS * HEADS * (HDIM / 2));  // 16.7M
  rope_kernel<<<rope_threads / 256, blk, 0, stream>>>(Qb, EMBED);       // Q
  rope_kernel<<<rope_threads / 256, blk, 0, stream>>>(KVb, 2 * EMBED);  // K (cols 0..1023)

  swa_attn_kernel<<<dim3(SEQ / WIN, HEADS, BATCH), blk, 0, stream>>>(Qb, KVb, Ob);

  out_proj_kernel<<<dim3(EMBED / 128, MS / 128), blk, 0, stream>>>(Ob, Wo, bo, out);
}
